// SelfAttention_50972671869478
// MI455X (gfx1250) — matrix operand order repaired and checked by reading, not compiled, not run on hardware
//
#include <hip/hip_runtime.h>
#include <hip/hip_bf16.h>

typedef __attribute__((ext_vector_type(16))) _Float16 v16h;
typedef __attribute__((ext_vector_type(8)))  _Float16 v8h;
typedef __attribute__((ext_vector_type(4)))  _Float16 v4h;
typedef __attribute__((ext_vector_type(8)))  float    v8f;

// ---------------------------------------------------------------------------
// CDNA5 async global->LDS copy (ASYNCcnt-tracked DMA, no VGPR staging).
// Per ISA 15.18.3: GLOBAL_LOAD_ASYNC_TO_LDS_B128, VDST = per-lane LDS byte
// address VGPR, VADDR = per-lane 64-bit global address.
// ---------------------------------------------------------------------------
__device__ inline void async_copy_b128(const _Float16* g, _Float16* l) {
    auto l3 = (__attribute__((address_space(3))) _Float16*)l;
    asm volatile("global_load_async_to_lds_b128 %0, %1, off"
                 :: "v"(l3), "v"(g) : "memory");
}
__device__ inline void wait_async_le12() {
#if __has_builtin(__builtin_amdgcn_s_wait_asynccnt)
    __builtin_amdgcn_s_wait_asynccnt(12);
#else
    asm volatile("s_wait_asynccnt 0xc" ::: "memory");
#endif
}
__device__ inline void wait_async_le0() {
#if __has_builtin(__builtin_amdgcn_s_wait_asynccnt)
    __builtin_amdgcn_s_wait_asynccnt(0);
#else
    asm volatile("s_wait_asynccnt 0x0" ::: "memory");
#endif
}

// ---------------------------------------------------------------------------
// fp32 -> fp16 conversion (vectorized x4)
// ---------------------------------------------------------------------------
__global__ void f32_to_f16x4(const float* __restrict__ in,
                             _Float16* __restrict__ out, long long n4) {
    long long i = (long long)blockIdx.x * blockDim.x + threadIdx.x;
    if (i >= n4) return;
    const float4 f = *(const float4*)(in + i * 4);
    v4h o = { (_Float16)f.x, (_Float16)f.y, (_Float16)f.z, (_Float16)f.w };
    *(v4h*)(out + i * 4) = o;
}

// ---------------------------------------------------------------------------
// Tiled transpose + fp32->fp16 convert for square n x n weights:
// out[N][K] = (fp16) in[K][N]. Done once per weight so all GEMM B-operands
// are [N][K] and hot loops need no transpose scatter.
// ---------------------------------------------------------------------------
__global__ __launch_bounds__(256)
void transpose_f32_to_f16(const float* __restrict__ in,
                          _Float16* __restrict__ out, int n) {
    __shared__ float tile[32][33];
    const int bx = blockIdx.x * 32, by = blockIdx.y * 32;
    const int tx = threadIdx.x, ty = threadIdx.y;   // block (32, 8)
#pragma unroll
    for (int j = 0; j < 32; j += 8)
        tile[ty + j][tx] = in[(long long)(by + ty + j) * n + bx + tx];
    __syncthreads();
#pragma unroll
    for (int j = 0; j < 32; j += 8)
        out[(long long)(bx + ty + j) * n + by + tx] = (_Float16)tile[tx][ty + j];
}

// ---------------------------------------------------------------------------
// WMMA fragment loads from LDS tiles (row-major [row][LS] halves).
// A 16x32 fp16 ISA layout: lane l (r=l%16, hi=l/16), row M=r, holds
//   K = hi*8 + [0..7] (VGPR0-3) and K = 16 + hi*8 + [0..7] (VGPR4-7).
// B 32x16 fp16 ISA layout: lane l, col N=r, holds K = hi*8 + [0..7] and K = 16 + hi*8 + [0..7].
// ---------------------------------------------------------------------------
__device__ inline v16h load_frag_A(const _Float16* s, int stride, int r, int hi) {
    const v8h lo = *(const v8h*)(s + r * stride + hi * 8);
    const v8h hh = *(const v8h*)(s + r * stride + 16 + hi * 8);
    return __builtin_shufflevector(lo, hh, 0,1,2,3,4,5,6,7,8,9,10,11,12,13,14,15);
}
__device__ inline v16h load_frag_B(const _Float16* s, int stride, int r, int hi) {
    const v8h lo = *(const v8h*)(s + r * stride + hi * 8);
    const v8h hh = *(const v8h*)(s + r * stride + 16 + hi * 8);
    return __builtin_shufflevector(lo, hh, 0,1,2,3,4,5,6,7,8,9,10,11,12,13,14,15);
}

// ---------------------------------------------------------------------------
// Batched fp16 GEMM: C = alpha * A * B^T   (B memory layout is [N][K])
//   A: [M][K] (lda = K-stride), B: [N][K] (ldb = K-stride)
//   TRANSC=false: C[m][n] (ldc = N-stride); TRANSC=true: C[n][m] (ldc = M-stride)
// Block: 256 threads = 8 waves; tile 128(M) x 256(N); K-stage 64.
// Wave grid 2(M) x 4(N); each wave computes 64x64 = 4x4 WMMA tiles
// (fragment reuse 4x: 16 ds_load_b128 per 16 WMMAs per 32-K substep).
// Double-buffered LDS fed by async global->LDS DMA: stage k+1 issued into
// buf^1, then s_wait_asynccnt<=12 drains stage k (in-order completion) while
// the 12 new ops stay in flight behind the WMMA work.
// ---------------------------------------------------------------------------
constexpr int BM = 128, BN = 256, TK = 64;
constexpr int LS = 80;              // LDS row stride in halves (160B, 16B-aligned)

template <typename OUTT, bool TRANSC>
__global__ __launch_bounds__(256, 2)
void gemm_wmma(const _Float16* __restrict__ A, const _Float16* __restrict__ Bm,
               OUTT* __restrict__ C,
               int M, int N, int K, int lda, int ldb, int ldc,
               long long strideA, long long strideB, long long strideC,
               float alpha) {
    __shared__ _Float16 sA[2][BM * LS];
    __shared__ _Float16 sB[2][BN * LS];

    const int tid  = threadIdx.x;
    const int wave = tid >> 5;
    const int lane = tid & 31;
    const int r    = lane & 15;
    const int hi   = lane >> 4;
    const int wm   = (wave >> 2) * 64;   // wave origin in M (0 or 64)
    const int wn   = (wave & 3) * 64;    // wave origin in N (0,64,128,192)

    const long long bz = blockIdx.z;
    const _Float16* Ab = A + bz * strideA;
    const _Float16* Bb = Bm + bz * strideB;
    OUTT*           Cb = C + bz * strideC;

    const int m0 = blockIdx.y * BM;
    const int n0 = blockIdx.x * BN;

    // staging coords: 256 threads; A: 4 async b128 (128 rows), B: 8 (256 rows)
    const int rowS = tid >> 3;           // 0..31 (+ i*32)
    const int g8   = (tid & 7) * 8;      // 16B group within the 64-half row

    const _Float16* gA = Ab + (long long)(m0 + rowS) * lda + g8;
    const _Float16* gB = Bb + (long long)(n0 + rowS) * ldb + g8;

    const v8f vzero = {0.f, 0.f, 0.f, 0.f, 0.f, 0.f, 0.f, 0.f};
    v8f acc[4][4];
#pragma unroll
    for (int mt = 0; mt < 4; ++mt)
#pragma unroll
        for (int nt = 0; nt < 4; ++nt) acc[mt][nt] = vzero;

    auto issue_stage = [&](int k0, int buf) {
#pragma unroll
        for (int i = 0; i < 4; ++i)
            async_copy_b128(gA + (long long)i * 32 * lda + k0,
                            &sA[buf][(rowS + i * 32) * LS + g8]);
#pragma unroll
        for (int i = 0; i < 8; ++i)
            async_copy_b128(gB + (long long)i * 32 * ldb + k0,
                            &sB[buf][(rowS + i * 32) * LS + g8]);
    };

    issue_stage(0, 0);
    int buf = 0;
    for (int k0 = 0; k0 < K; k0 += TK) {
        const bool more = (k0 + TK) < K;
        if (more) {
            issue_stage(k0 + TK, buf ^ 1);   // safe: buf^1 readers done (prev barrier)
            wait_async_le12();               // stage k0 landed; 12 new still in flight
        } else {
            wait_async_le0();
        }
        __syncthreads();

        // ---- compute: 2 substeps x (4 A-frags, 4 B-frags, 16 WMMAs) ----
#pragma unroll
        for (int ks = 0; ks < TK; ks += 32) {
            v16h af[4];
#pragma unroll
            for (int mt = 0; mt < 4; ++mt)
                af[mt] = load_frag_A(&sA[buf][(wm + mt * 16) * LS + ks], LS, r, hi);
#pragma unroll
            for (int nt = 0; nt < 4; ++nt) {
                const v16h bf = load_frag_B(&sB[buf][(wn + nt * 16) * LS + ks], LS, r, hi);
#pragma unroll
                for (int mt = 0; mt < 4; ++mt)
                    acc[mt][nt] = __builtin_amdgcn_wmma_f32_16x16x32_f16(
                        false, af[mt], false, bf, (short)0, acc[mt][nt], false, false);
            }
        }
        __syncthreads();
        buf ^= 1;
    }

    // ---- epilogue ----
#pragma unroll
    for (int mt = 0; mt < 4; ++mt)
#pragma unroll
        for (int nt = 0; nt < 4; ++nt) {
            const int mb = m0 + wm + mt * 16;
            const int nb = n0 + wn + nt * 16;
            if constexpr (TRANSC) {
                // C[n][m]: per lane one contiguous 16B v8h store
                v8h o;
#pragma unroll
                for (int i = 0; i < 8; ++i)
                    o[i] = (_Float16)(alpha * acc[mt][nt][i]);
                *(v8h*)((_Float16*)Cb + (long long)(nb + r) * ldc + mb + hi * 8) = o;
            } else {
                // C[m][n]: element i lives at row (hi*8+i), col r
#pragma unroll
                for (int i = 0; i < 8; ++i) {
                    const float v = alpha * acc[mt][nt][i];
                    Cb[(long long)(mb + hi * 8 + i) * ldc + nb + r] = (OUTT)v;
                }
            }
        }
}

// ---------------------------------------------------------------------------
// Row softmax over fp16 logits, in place. One block (256 threads) per row of
// 2048; each thread owns 8 contiguous columns.
// ---------------------------------------------------------------------------
__global__ __launch_bounds__(256)
void softmax_rows(_Float16* __restrict__ P, int ncols) {
    __shared__ float red[256];
    const int tid = threadIdx.x;
    _Float16* p = P + (long long)blockIdx.x * ncols;

    const v8h d = *(const v8h*)(p + tid * 8);
    float v[8];
    float mx = -3.0e38f;
#pragma unroll
    for (int j = 0; j < 8; ++j) { v[j] = (float)d[j]; mx = fmaxf(mx, v[j]); }

    red[tid] = mx;
    __syncthreads();
    for (int s = 128; s > 0; s >>= 1) {
        if (tid < s) red[tid] = fmaxf(red[tid], red[tid + s]);
        __syncthreads();
    }
    mx = red[0];
    __syncthreads();

    float sum = 0.f;
#pragma unroll
    for (int j = 0; j < 8; ++j) { v[j] = __expf(v[j] - mx); sum += v[j]; }

    red[tid] = sum;
    __syncthreads();
    for (int s = 128; s > 0; s >>= 1) {
        if (tid < s) red[tid] += red[tid + s];
        __syncthreads();
    }
    const float inv = 1.0f / red[0];

    v8h o;
#pragma unroll
    for (int j = 0; j < 8; ++j) o[j] = (_Float16)(v[j] * inv);
    *(v8h*)(p + tid * 8) = o;
}

// ---------------------------------------------------------------------------
// Launcher. Inputs: x[B,S,D], Wq, Wk, Wv, Wo (fp32). Output fp32 [B,S,D].
// Workspace (fp16 elems, ~120 MiB):
//   xh[8192x1024] | Wq/Wk/Wv/Wo^T [1024x1024] | Q,K [8192x1024] |
//   V^T [1024x8192] | P [4x2048x2048] | Y [8192x1024]
// ---------------------------------------------------------------------------
extern "C" void kernel_launch(void* const* d_in, const int* in_sizes, int n_in,
                              void* d_out, int out_size, void* d_ws, size_t ws_size,
                              hipStream_t stream) {
    constexpr int  Bb = 4, S = 2048, D = 1024;
    constexpr long long XE = (long long)Bb * S * D;
    constexpr long long WE = (long long)D * D;
    constexpr long long PE = (long long)Bb * S * S;

    const float* x  = (const float*)d_in[0];
    const float* Wq = (const float*)d_in[1];
    const float* Wk = (const float*)d_in[2];
    const float* Wv = (const float*)d_in[3];
    const float* Wo = (const float*)d_in[4];
    float* out = (float*)d_out;

    _Float16* xh  = (_Float16*)d_ws;
    _Float16* Wqt = xh + XE;
    _Float16* Wkt = Wqt + WE;
    _Float16* Wvt = Wkt + WE;
    _Float16* Wot = Wvt + WE;
    _Float16* Qh  = Wot + WE;
    _Float16* Kh  = Qh + XE;
    _Float16* Vt  = Kh + XE;       // [D][Bb*S] = [1024][8192]
    _Float16* P   = Vt + XE;
    _Float16* Yh  = P + PE;

    // staging: x elementwise, weights transposed to [N][K]
    f32_to_f16x4<<<dim3((unsigned)(XE / 4 / 256)), 256, 0, stream>>>(x, xh, XE / 4);
    const dim3 tb(32, 8), tg(D / 32, D / 32);
    transpose_f32_to_f16<<<tg, tb, 0, stream>>>(Wq, Wqt, D);
    transpose_f32_to_f16<<<tg, tb, 0, stream>>>(Wk, Wkt, D);
    transpose_f32_to_f16<<<tg, tb, 0, stream>>>(Wv, Wvt, D);
    transpose_f32_to_f16<<<tg, tb, 0, stream>>>(Wo, Wot, D);

    const dim3 blk(256);
    const float qscale = 0.03125f;   // 1/sqrt(D), folded into Q

    // Projections [M=8192, N=1024, K=1024]; V stored transposed [1024][8192]
    gemm_wmma<_Float16, false><<<dim3(D / BN, (Bb * S) / BM, 1), blk, 0, stream>>>(
        xh, Wqt, Qh, Bb * S, D, D, D, D, D, 0, 0, 0, qscale);
    gemm_wmma<_Float16, false><<<dim3(D / BN, (Bb * S) / BM, 1), blk, 0, stream>>>(
        xh, Wkt, Kh, Bb * S, D, D, D, D, D, 0, 0, 0, 1.0f);
    gemm_wmma<_Float16, true><<<dim3(D / BN, (Bb * S) / BM, 1), blk, 0, stream>>>(
        xh, Wvt, Vt, Bb * S, D, D, D, D, Bb * S, 0, 0, 0, 1.0f);

    // Logits: P_b = Q_b K_b^T   [per batch M=2048, N=2048, K=1024]
    gemm_wmma<_Float16, false><<<dim3(S / BN, S / BM, Bb), blk, 0, stream>>>(
        Qh, Kh, P, S, S, D, D, D, S,
        (long long)S * D, (long long)S * D, (long long)S * S, 1.0f);

    softmax_rows<<<dim3(Bb * S), 256, 0, stream>>>(P, S);

    // Y_b = P_b V_b  (B = V^T slice: ldb = Bb*S, batch offset = b*S)
    gemm_wmma<_Float16, false><<<dim3(D / BN, S / BM, Bb), blk, 0, stream>>>(
        P, Vt, Yh, S, D, S, S, Bb * S, D,
        (long long)S * S, (long long)S, (long long)S * D, 1.0f);

    // out = Y Wo (fp32 output)  [M=8192, N=1024, K=1024]
    gemm_wmma<float, false><<<dim3(D / BN, (Bb * S) / BM, 1), blk, 0, stream>>>(
        Yh, Wot, out, Bb * S, D, D, D, D, D, 0, 0, 0, 1.0f);
}
